// ContextualModule_48868137894456
// MI455X (gfx1250) — matrix operand order repaired and checked by reading, not compiled, not run on hardware
//
#include <hip/hip_runtime.h>
#include <math.h>

// ---------------------------------------------------------------------------
// ContextualModule for MI455X (gfx1250, wave32, WMMA f16/f32-acc).
// 128x64 tiles, register-staged double-buffered LDS pipeline, 1 barrier/step.
// f32 -> f16 staging via v_cvt_pkrtz_f16_f32 (1 VALU per 2 elements).
// ---------------------------------------------------------------------------

#define NB_   8
#define CC    512
#define HH    66
#define WW    90
#define HWP   (HH * WW)           // 5940
#define NBIG  (NB_ * CC * HWP)    // 24330240

typedef __attribute__((ext_vector_type(16))) _Float16 v16h;
typedef __attribute__((ext_vector_type(2)))  __fp16   h2pk;   // cvt_pkrtz result type
typedef __attribute__((ext_vector_type(8)))  float    v8f;

#define BM  128
#define BN  64
#define BK  32
#define LDT 40   // LDS row stride in f16 elements (80B: 16B-aligned, bank-skewed)

// pack two f32 -> packed f16x2 (lo = f0), single v_cvt_pkrtz_f16_f32
__device__ __forceinline__ unsigned pack_f16x2(float f0, float f1) {
  union { h2pk v; unsigned u; } r;
  r.v = __builtin_amdgcn_cvt_pkrtz(f0, f1);
  return r.u;
}
__device__ __forceinline__ unsigned short f2h(float f) {
  union { _Float16 h; unsigned short u; } r;
  r.h = (_Float16)f;
  return r.u;
}

union F4 { float4 v; float f[4]; };
union FragU { uint4 u[2]; v16h v; };

// A fragment (16x32 f16): lanes 0-15 K 0..7 & 16..23; lanes 16-31 K 8..15 & 24..31
__device__ __forceinline__ v16h load_a_frag(const unsigned short* As, int row, int lhalf) {
  const unsigned short* rp = As + row * LDT;
  const int kb = lhalf * 8;
  FragU f;
  f.u[0] = *(const uint4*)(rp + kb);
  f.u[1] = *(const uint4*)(rp + kb + 16);
  return f.v;
}
// B fragment (32x16 f16): lane col = lane%16; lanes 0-15 K 0..7 & 16..23; lanes 16-31 K 8..15 & 24..31
__device__ __forceinline__ v16h load_b_frag(const unsigned short* Bs, int col, int lhalf) {
  const unsigned short* rp = Bs + col * LDT;
  const int kb = lhalf * 8;
  FragU f;
  f.u[0] = *(const uint4*)(rp + kb);
  f.u[1] = *(const uint4*)(rp + kb + 16);
  return f.v;
}
__device__ __forceinline__ v8f wmma_f16(v16h a, v16h b, v8f c) {
  return __builtin_amdgcn_wmma_f32_16x16x32_f16(false, a, false, b, (short)0, c, false, false);
}

// bilinear upsample (half-pixel) of tiny sz x sz map to 66x90 at (y, x)
__device__ __forceinline__ float bilin66x90(const float* q, int sz, int y, int x) {
  float fy = (y + 0.5f) * (float)sz * (1.0f / 66.0f) - 0.5f;
  float fx = (x + 0.5f) * (float)sz * (1.0f / 90.0f) - 0.5f;
  float yf = floorf(fy), xf = floorf(fx);
  int y0 = (int)yf, x0 = (int)xf;
  float wy = fy - yf, wx = fx - xf;
  int y0c = min(max(y0, 0), sz - 1);
  int y1c = min(max(y0 + 1, 0), sz - 1);
  int x0c = min(max(x0, 0), sz - 1);
  int x1c = min(max(x0 + 1, 0), sz - 1);
  float v00 = q[y0c * sz + x0c], v01 = q[y0c * sz + x1c];
  float v10 = q[y1c * sz + x0c], v11 = q[y1c * sz + x1c];
  float v0 = v00 + (v01 - v00) * wx;
  float v1 = v10 + (v11 - v10) * wx;
  return v0 + (v1 - v0) * wy;
}

// ---------------------------------------------------------------------------
// Batched GEMM (1x1 conv): Y[n,m,p] = sum_k A[m,k] * B[n,k,p] (+bias, epi)
// BMODE: 0 plain, 1 B = feats - up(q), 2 virtual concat (k<Ksplit -> B0 else B1)
// EPI:   0 bias, 1 bias+sigmoid -> num += up*s, den += s, 2 bias+relu
// VECB:  1 = stride HWP, vector-staged B; 0 = runtime stride, scalar B
// ---------------------------------------------------------------------------
struct GemmArgs {
  const float* A;
  const float* B0;
  const float* B1;
  const float* bias;
  float*       out;
  const float* q;
  float*       num;
  float*       den;
  int M, Ntot, Ksplit, sz, bStrideK;
  long bBatch, oBatch;
};

template<int BMODE, int EPI, int VECB, int KT>
__global__ __launch_bounds__(256) void gemm_wmma(GemmArgs g) {
  __shared__ __align__(16) unsigned short As[2][BM * LDT];
  __shared__ __align__(16) unsigned short Bs[2][BN * LDT];
  const int nB   = blockIdx.z;
  const int m0   = blockIdx.y * BM;
  const int p0   = blockIdx.x * BN;
  const int tid  = threadIdx.x;
  const int lane = tid & 31, wave = tid >> 5;
  const int wm = wave >> 1, wn = wave & 1;            // 4x2 wave grid, 32x32 per wave
  const int lhalf = lane >> 4, lmod = lane & 15;

  const float* Abase = g.A + (long)(m0 + (tid >> 3)) * KT + (tid & 7) * 4;
  const float* B0 = g.B0 + (long)nB * g.bBatch;
  const float* B1b = (BMODE == 2) ? (g.B1 + (long)nB * g.bBatch) : nullptr;

  const bool fullN = (p0 + BN <= g.Ntot);
  const int  bk = (tid >> 4) * 2;        // k pair base within tile (VECB)
  const int  bn = (tid & 15) * 4;        // n quad base within tile (VECB)

  F4 aR[4];
  F4 bR0, bR1;      // VECB regs
  float bS[8];      // scalar regs

  v8f acc00 = {}, acc01 = {}, acc10 = {}, acc11 = {};

  auto loadA = [&](int k0) {
#pragma unroll
    for (int i = 0; i < 4; ++i)
      aR[i].v = *(const float4*)(Abase + (long)i * 32 * KT + k0);
  };

  auto loadB = [&](int k0) {
    if (VECB) {
      const int kg = k0 + bk;
      if (fullN) {
        const float *src0, *src1;
        if (BMODE == 2) {
          src0 = (kg     < g.Ksplit) ? (B0 + (long)kg * HWP)
                                     : (B1b + (long)(kg - g.Ksplit) * HWP);
          src1 = (kg + 1 < g.Ksplit) ? (B0 + (long)(kg + 1) * HWP)
                                     : (B1b + (long)(kg + 1 - g.Ksplit) * HWP);
        } else {
          src0 = B0 + (long)kg * HWP;
          src1 = src0 + HWP;
        }
        const int p = p0 + bn;
        bR0.v = *(const float4*)(src0 + p);
        bR1.v = *(const float4*)(src1 + p);
        if (BMODE == 1) {
          const int szsq = g.sz * g.sz;
          const float* q0 = g.q + ((long)nB * CC + kg) * szsq;
          const float* q1 = q0 + szsq;
#pragma unroll
          for (int j = 0; j < 4; ++j) {
            int pj = p + j;
            int y = pj / WW, x = pj - y * WW;
            bR0.f[j] -= bilin66x90(q0, g.sz, y, x);
            bR1.f[j] -= bilin66x90(q1, g.sz, y, x);
          }
        }
      } else {
#pragma unroll
        for (int j = 0; j < 4; ++j) {
          int p = p0 + bn + j;
          float v0 = 0.0f, v1 = 0.0f;
          if (p < g.Ntot) {
            if (BMODE == 2) {
              v0 = (kg     < g.Ksplit) ? B0[(long)kg * HWP + p]
                                       : B1b[(long)(kg - g.Ksplit) * HWP + p];
              v1 = (kg + 1 < g.Ksplit) ? B0[(long)(kg + 1) * HWP + p]
                                       : B1b[(long)(kg + 1 - g.Ksplit) * HWP + p];
            } else {
              v0 = B0[(long)kg * HWP + p];
              v1 = B0[(long)(kg + 1) * HWP + p];
              if (BMODE == 1) {
                const int szsq = g.sz * g.sz;
                const float* q0 = g.q + ((long)nB * CC + kg) * szsq;
                int y = p / WW, x = p - y * WW;
                v0 -= bilin66x90(q0, g.sz, y, x);
                v1 -= bilin66x90(q0 + szsq, g.sz, y, x);
              }
            }
          }
          bR0.f[j] = v0;
          bR1.f[j] = v1;
        }
      }
    } else {
#pragma unroll
      for (int i = 0; i < 8; ++i) {
        int idx = tid + i * 256;
        int kk = idx >> 6, nn = idx & 63;
        int p = p0 + nn;
        bS[i] = (p < g.Ntot) ? B0[(long)(k0 + kk) * g.bStrideK + p] : 0.0f;
      }
    }
  };

  auto storeA = [&](int buf) {
#pragma unroll
    for (int i = 0; i < 4; ++i) {
      int r = (tid >> 3) + i * 32;
      int c = (tid & 7) * 4;
      unsigned lo = pack_f16x2(aR[i].f[0], aR[i].f[1]);
      unsigned hi = pack_f16x2(aR[i].f[2], aR[i].f[3]);
      *(uint2*)&As[buf][r * LDT + c] = make_uint2(lo, hi);
    }
  };

  auto storeB = [&](int buf) {
    if (VECB) {
#pragma unroll
      for (int j = 0; j < 4; ++j)
        *(unsigned*)&Bs[buf][(bn + j) * LDT + bk] = pack_f16x2(bR0.f[j], bR1.f[j]);
    } else {
#pragma unroll
      for (int i = 0; i < 8; ++i) {
        int idx = tid + i * 256;
        int kk = idx >> 6, nn = idx & 63;
        Bs[buf][nn * LDT + kk] = f2h(bS[i]);
      }
    }
  };

  auto compute = [&](int buf) {
    v16h a0 = load_a_frag(As[buf], wm * 32 + lmod, lhalf);
    v16h a1 = load_a_frag(As[buf], wm * 32 + 16 + lmod, lhalf);
    v16h b0 = load_b_frag(Bs[buf], wn * 32 + lmod, lhalf);
    v16h b1 = load_b_frag(Bs[buf], wn * 32 + 16 + lmod, lhalf);
    acc00 = wmma_f16(a0, b0, acc00);
    acc01 = wmma_f16(a0, b1, acc01);
    acc10 = wmma_f16(a1, b0, acc10);
    acc11 = wmma_f16(a1, b1, acc11);
  };

  const int nK = KT / BK;
  loadA(0); loadB(0);
  storeA(0); storeB(0);
  __syncthreads();
  for (int ks = 0; ks < nK; ++ks) {
    const bool more = (ks + 1 < nK);
    if (more) {
      loadA((ks + 1) * BK);
      loadB((ks + 1) * BK);
      if (VECB && fullN && BMODE != 2 && (ks + 4 < nK))
        __builtin_prefetch(B0 + (long)((ks + 4) * BK + bk) * HWP + p0 + bn, 0, 1);
    }
    compute(ks & 1);
    if (more) { storeA((ks + 1) & 1); storeB((ks + 1) & 1); }
    __syncthreads();
  }

  float* outB = g.out ? (g.out + (long)nB * g.oBatch) : nullptr;
  v8f accs[2][2] = {{acc00, acc01}, {acc10, acc11}};
#pragma unroll
  for (int sub = 0; sub < 2; ++sub) {
#pragma unroll
    for (int cg = 0; cg < 2; ++cg) {
      int p = p0 + wn * 32 + cg * 16 + lmod;
      if (p >= g.Ntot) continue;
      v8f a = accs[sub][cg];
#pragma unroll
      for (int r = 0; r < 8; ++r) {
        int m = m0 + wm * 32 + sub * 16 + lhalf * 8 + r;
        float v = a[r] + (g.bias ? g.bias[m] : 0.0f);
        if (EPI == 1) {
          float s = 1.0f / (1.0f + expf(-v));
          int y = p / WW, x = p - y * WW;
          float up = bilin66x90(g.q + ((long)nB * CC + m) * (g.sz * g.sz), g.sz, y, x);
          long idx = ((long)nB * CC + m) * HWP + p;
          g.num[idx] += up * s;
          g.den[idx] += s;
        } else {
          if (EPI == 2) v = fmaxf(v, 0.0f);
          outB[(long)m * g.Ntot + p] = v;
        }
      }
    }
  }
}

// ---------------------------------------------------------------------------
// 3x3 conv: 9 taps x 16 K-steps of WMMA, weights pre-repacked to [tap][co][ci]
// ---------------------------------------------------------------------------
struct Conv3Args {
  const float* wt;    // repacked [9][C][C]
  const float* x;     // [NB_, C, inH, inW]
  const float* bias;  // [C]
  float*       out;   // [NB_, C, outH, outW]
  int inH, inW, outH, outW, stride, pad;
};

__global__ __launch_bounds__(256) void conv3_wmma(Conv3Args c) {
  __shared__ __align__(16) unsigned short As[2][BM * LDT];
  __shared__ __align__(16) unsigned short Bs[2][BN * LDT];
  const int nB   = blockIdx.z;
  const int m0   = blockIdx.y * BM;
  const int p0   = blockIdx.x * BN;
  const int P    = c.outH * c.outW;
  const int tid  = threadIdx.x;
  const int lane = tid & 31, wave = tid >> 5;
  const int wm = wave >> 1, wn = wave & 1;
  const int lhalf = lane >> 4, lmod = lane & 15;

  F4 aR[4];
  float bS[8];
  v8f acc00 = {}, acc01 = {}, acc10 = {}, acc11 = {};

  auto loadA = [&](int s) {
    int tap = s >> 4, ks = s & 15;
    const float* base = c.wt + (long)tap * CC * CC
                      + (long)(m0 + (tid >> 3)) * CC + (tid & 7) * 4 + ks * BK;
#pragma unroll
    for (int i = 0; i < 4; ++i)
      aR[i].v = *(const float4*)(base + (long)i * 32 * CC);
  };
  auto loadB = [&](int s) {
    int tap = s >> 4, ks = s & 15;
    int dy = tap / 3, dx = tap % 3;
#pragma unroll
    for (int i = 0; i < 8; ++i) {
      int idx = tid + i * 256;
      int kk = idx >> 6, nn = idx & 63;
      int p = p0 + nn;
      float v = 0.0f;
      if (p < P) {
        int oy = p / c.outW, ox = p - oy * c.outW;
        int iy = oy * c.stride + dy - c.pad;
        int ix = ox * c.stride + dx - c.pad;
        if (iy >= 0 && iy < c.inH && ix >= 0 && ix < c.inW)
          v = c.x[(((long)nB * CC + ks * BK + kk) * c.inH + iy) * c.inW + ix];
      }
      bS[i] = v;
    }
  };
  auto storeA = [&](int buf) {
#pragma unroll
    for (int i = 0; i < 4; ++i) {
      int r = (tid >> 3) + i * 32;
      int col = (tid & 7) * 4;
      unsigned lo = pack_f16x2(aR[i].f[0], aR[i].f[1]);
      unsigned hi = pack_f16x2(aR[i].f[2], aR[i].f[3]);
      *(uint2*)&As[buf][r * LDT + col] = make_uint2(lo, hi);
    }
  };
  auto storeB = [&](int buf) {
#pragma unroll
    for (int i = 0; i < 8; ++i) {
      int idx = tid + i * 256;
      int kk = idx >> 6, nn = idx & 63;
      Bs[buf][nn * LDT + kk] = f2h(bS[i]);
    }
  };
  auto compute = [&](int buf) {
    v16h a0 = load_a_frag(As[buf], wm * 32 + lmod, lhalf);
    v16h a1 = load_a_frag(As[buf], wm * 32 + 16 + lmod, lhalf);
    v16h b0 = load_b_frag(Bs[buf], wn * 32 + lmod, lhalf);
    v16h b1 = load_b_frag(Bs[buf], wn * 32 + 16 + lmod, lhalf);
    acc00 = wmma_f16(a0, b0, acc00);
    acc01 = wmma_f16(a0, b1, acc01);
    acc10 = wmma_f16(a1, b0, acc10);
    acc11 = wmma_f16(a1, b1, acc11);
  };

  const int nStep = 9 * (CC / BK);   // 144
  loadA(0); loadB(0);
  storeA(0); storeB(0);
  __syncthreads();
  for (int s = 0; s < nStep; ++s) {
    const bool more = (s + 1 < nStep);
    if (more) { loadA(s + 1); loadB(s + 1); }
    compute(s & 1);
    if (more) { storeA((s + 1) & 1); storeB((s + 1) & 1); }
    __syncthreads();
  }

  float* outB = c.out + (long)nB * CC * P;
  v8f accs[2][2] = {{acc00, acc01}, {acc10, acc11}};
#pragma unroll
  for (int sub = 0; sub < 2; ++sub) {
#pragma unroll
    for (int cg = 0; cg < 2; ++cg) {
      int p = p0 + wn * 32 + cg * 16 + lmod;
      if (p >= P) continue;
      v8f a = accs[sub][cg];
#pragma unroll
      for (int r = 0; r < 8; ++r) {
        int m = m0 + wm * 32 + sub * 16 + lhalf * 8 + r;
        outB[(long)m * P + p] = a[r] + c.bias[m];
      }
    }
  }
}

// ---------------------------------------------------------------------------
// repack OIHW 3x3 weights -> [tap][co][ci]
__global__ void repack_w9(const float* __restrict__ w, float* __restrict__ wt, long total) {
  for (long j = blockIdx.x * 256L + threadIdx.x; j < total; j += (long)gridDim.x * 256L) {
    long tap = j / (CC * CC);
    long rem = j - tap * (CC * CC);
    wt[j] = w[rem * 9 + tap];
  }
}

// adaptive avg pools to 1x1, 2x2, 3x3, 6x6 (exact block means) in one pass
__global__ __launch_bounds__(256) void pool_kernel(const float* __restrict__ feats,
                                                   float* __restrict__ pool) {
  __shared__ float bins[50];
  const int bid = blockIdx.x;            // n*C + c
  const int tid = threadIdx.x;
  if (tid < 50) bins[tid] = 0.0f;
  __syncthreads();
  const float* src = feats + (long)bid * HWP;
  float s1 = 0.0f;
  for (int p = tid; p < HWP; p += 256) {
    float v = src[p];
    int y = p / WW, x = p - y * WW;
    s1 += v;
    atomicAdd(&bins[1  + (y / 33) * 2 + (x / 45)], v);
    atomicAdd(&bins[5  + (y / 22) * 3 + (x / 30)], v);
    atomicAdd(&bins[14 + (y / 11) * 6 + (x / 15)], v);
  }
  atomicAdd(&bins[0], s1);
  __syncthreads();
  if (tid == 0)               pool[0     + bid]                   = bins[0]   * (1.0f / 5940.0f);
  if (tid >= 1  && tid < 5)   pool[4096  + bid * 4 + (tid - 1)]   = bins[tid] * (1.0f / (33.0f * 45.0f));
  if (tid >= 5  && tid < 14)  pool[20480 + bid * 9 + (tid - 5)]   = bins[tid] * (1.0f / (22.0f * 30.0f));
  if (tid >= 14 && tid < 50)  pool[57344 + bid * 36 + (tid - 14)] = bins[tid] * (1.0f / (11.0f * 15.0f));
}

__global__ void fill_kernel(float* p, float v, long n) {
  for (long i = blockIdx.x * 256L + threadIdx.x; i < n; i += (long)gridDim.x * 256L)
    p[i] = v;
}

__global__ void ratio_kernel(const float* __restrict__ num, const float* __restrict__ den,
                             float* __restrict__ out, long n) {
  for (long i = blockIdx.x * 256L + threadIdx.x; i < n; i += (long)gridDim.x * 256L)
    out[i] = num[i] / den[i];
}

__global__ void maxpool_kernel(const float* __restrict__ in, float* __restrict__ out,
                               int inH, int inW, int outH, int outW, int k, int s, long total) {
  const int P = outH * outW;
  for (long i = blockIdx.x * 256L + threadIdx.x; i < total; i += (long)gridDim.x * 256L) {
    long nc = i / P;
    int rem = (int)(i - nc * P);
    int oy = rem / outW, ox = rem - oy * outW;
    const float* src = in + nc * (long)(inH * inW);
    float m = -3.402823466e38f;
    for (int ky = 0; ky < k; ++ky)
      for (int kx = 0; kx < k; ++kx)
        m = fmaxf(m, src[(oy * s + ky) * inW + (ox * s + kx)]);
    out[i] = m;
  }
}

// per-(n,c) dynamic depthwise 5x7 conv, pad (2,3)
__global__ __launch_bounds__(256) void dyn_dw(const float* __restrict__ inp,
                                              const float* __restrict__ theta,
                                              float* __restrict__ out) {
  __shared__ float th[35];
  const int nc = blockIdx.y;
  if (threadIdx.x < 35) th[threadIdx.x] = theta[(long)nc * 35 + threadIdx.x];
  __syncthreads();
  const int p = blockIdx.x * 256 + threadIdx.x;
  if (p >= HWP) return;
  const int y = p / WW, x = p - (p / WW) * WW;
  const float* src = inp + (long)nc * HWP;
  float acc = 0.0f;
#pragma unroll
  for (int ky = 0; ky < 5; ++ky) {
    int iy = y + ky - 2;
    if (iy < 0 || iy >= HH) continue;
#pragma unroll
    for (int kx = 0; kx < 7; ++kx) {
      int ix = x + kx - 3;
      if (ix < 0 || ix >= WW) continue;
      acc += src[iy * WW + ix] * th[ky * 7 + kx];
    }
  }
  out[(long)nc * HWP + p] = acc;
}

// ---------------------------------------------------------------------------
extern "C" void kernel_launch(void* const* d_in, const int* in_sizes, int n_in,
                              void* d_out, int out_size, void* d_ws, size_t ws_size,
                              hipStream_t stream) {
  (void)in_sizes; (void)n_in; (void)out_size; (void)ws_size;
  const float* feats   = (const float*)d_in[0];
  const float* scale_w = (const float*)d_in[1];
  const float* wn_w    = (const float*)d_in[2];
  const float* wn_b    = (const float*)d_in[3];
  const float* pa_w1   = (const float*)d_in[4];
  const float* pa_b1   = (const float*)d_in[5];
  const float* pa_w2   = (const float*)d_in[6];
  const float* pa_b2   = (const float*)d_in[7];
  const float* pa_w3   = (const float*)d_in[8];
  const float* pa_b3   = (const float*)d_in[9];
  const float* u_w     = (const float*)d_in[10];
  const float* u_b     = (const float*)d_in[11];
  const float* v_w     = (const float*)d_in[12];
  const float* v_b     = (const float*)d_in[13];
  const float* bott_w  = (const float*)d_in[14];
  const float* bott_b  = (const float*)d_in[15];
  float* outp = (float*)d_out;

  char* base = (char*)d_ws;
  size_t off = 0;
  auto alloc = [&](size_t nf) {
    size_t o = off;
    off += (nf * sizeof(float) + 255) & ~(size_t)255;
    return (float*)(base + o);
  };
  float* pool    = alloc(204800);
  float* qbuf    = alloc(204800);
  float* num     = alloc(NBIG);
  float* den     = alloc(NBIG);
  float* parsing = alloc(NBIG);
  float* pose    = alloc(NBIG);
  float* t1      = alloc((size_t)NB_ * CC * 32 * 44);
  float* t1p     = alloc((size_t)NB_ * CC * 16 * 22);
  float* t2      = alloc((size_t)NB_ * CC * 16 * 22);
  float* t2p     = alloc((size_t)NB_ * CC * 5 * 7);
  float* theta   = alloc((size_t)NB_ * CC * 5 * 7);
  float* wt1     = alloc((size_t)9 * CC * CC);
  float* wt2     = alloc((size_t)9 * CC * CC);
  float* wt3     = alloc((size_t)9 * CC * CC);
  float* inp     = num;   // reuse: num/den dead after `parsing`
  float* posein  = den;

  const int  sz[4]      = {1, 2, 3, 6};
  const int  szq[4]     = {1, 4, 9, 36};
  const long poolOff[4] = {0, 4096, 20480, 57344};
  const long W9 = (long)9 * CC * CC;

  // weight repacks for the 3x3 convs (once per launch, tiny)
  repack_w9<<<dim3(2048), dim3(256), 0, stream>>>(pa_w1, wt1, W9);
  repack_w9<<<dim3(2048), dim3(256), 0, stream>>>(pa_w2, wt2, W9);
  repack_w9<<<dim3(2048), dim3(256), 0, stream>>>(pa_w3, wt3, W9);

  // 1) adaptive avg pools
  pool_kernel<<<dim3(NB_ * CC), dim3(256), 0, stream>>>(feats, pool);

  // 2) per-scale 1x1 conv on pooled maps (no bias)
  for (int i = 0; i < 4; ++i) {
    GemmArgs g{};
    g.A = scale_w + (size_t)i * CC * CC;
    g.B0 = pool + poolOff[i];
    g.bias = nullptr;
    g.out = qbuf + poolOff[i];
    g.M = CC; g.Ntot = szq[i]; g.sz = sz[i]; g.bStrideK = szq[i];
    g.bBatch = (long)CC * szq[i]; g.oBatch = (long)CC * szq[i];
    gemm_wmma<0, 0, 0, CC><<<dim3(1, CC / BM, NB_), dim3(256), 0, stream>>>(g);
  }

  // 3) zero num/den
  fill_kernel<<<dim3(4096), dim3(256), 0, stream>>>(num, 0.0f, (long)NBIG);
  fill_kernel<<<dim3(4096), dim3(256), 0, stream>>>(den, 0.0f, (long)NBIG);

  // 4) wn branch fused: num += up_i * sigmoid(.), den += sigmoid(.)
  for (int i = 0; i < 4; ++i) {
    GemmArgs g{};
    g.A = wn_w; g.B0 = feats; g.bias = wn_b; g.out = nullptr;
    g.q = qbuf + poolOff[i]; g.num = num; g.den = den;
    g.M = CC; g.Ntot = HWP; g.sz = sz[i]; g.bStrideK = HWP;
    g.bBatch = (long)CC * HWP; g.oBatch = (long)CC * HWP;
    gemm_wmma<1, 1, 1, CC><<<dim3((HWP + BN - 1) / BN, CC / BM, NB_), dim3(256), 0, stream>>>(g);
  }

  // 5) parsing = num / den
  ratio_kernel<<<dim3(4096), dim3(256), 0, stream>>>(num, den, parsing, (long)NBIG);

  // 6) param_adapter
  Conv3Args c1{wt1, parsing, pa_b1, t1, 66, 90, 32, 44, 2, 0};
  conv3_wmma<<<dim3((32 * 44 + BN - 1) / BN, CC / BM, NB_), dim3(256), 0, stream>>>(c1);
  maxpool_kernel<<<dim3(2048), dim3(256), 0, stream>>>(t1, t1p, 32, 44, 16, 22, 2, 2,
                                                       (long)NB_ * CC * 16 * 22);
  Conv3Args c2{wt2, t1p, pa_b2, t2, 16, 22, 16, 22, 1, 1};
  conv3_wmma<<<dim3((16 * 22 + BN - 1) / BN, CC / BM, NB_), dim3(256), 0, stream>>>(c2);
  maxpool_kernel<<<dim3(1024), dim3(256), 0, stream>>>(t2, t2p, 16, 22, 5, 7, 3, 3,
                                                       (long)NB_ * CC * 5 * 7);
  Conv3Args c3{wt3, t2p, pa_b3, theta, 5, 7, 5, 7, 1, 1};
  conv3_wmma<<<dim3(1, CC / BM, NB_), dim3(256), 0, stream>>>(c3);

  // 7) inp = u_w x feats + u_b
  {
    GemmArgs g{};
    g.A = u_w; g.B0 = feats; g.bias = u_b; g.out = inp;
    g.M = CC; g.Ntot = HWP; g.bStrideK = HWP;
    g.bBatch = (long)CC * HWP; g.oBatch = (long)CC * HWP;
    gemm_wmma<0, 0, 1, CC><<<dim3((HWP + BN - 1) / BN, CC / BM, NB_), dim3(256), 0, stream>>>(g);
  }

  // 8) dynamic depthwise 5x7
  dyn_dw<<<dim3((HWP + 255) / 256, NB_ * CC), dim3(256), 0, stream>>>(inp, theta, posein);

  // 9) pose = relu(v_w x posein + v_b)
  {
    GemmArgs g{};
    g.A = v_w; g.B0 = posein; g.bias = v_b; g.out = pose;
    g.M = CC; g.Ntot = HWP; g.bStrideK = HWP;
    g.bBatch = (long)CC * HWP; g.oBatch = (long)CC * HWP;
    gemm_wmma<0, 2, 1, CC><<<dim3((HWP + BN - 1) / BN, CC / BM, NB_), dim3(256), 0, stream>>>(g);
  }

  // 10) out = relu(bott_w x concat(pose, feats) + bott_b)
  {
    GemmArgs g{};
    g.A = bott_w; g.B0 = pose; g.B1 = feats; g.bias = bott_b; g.out = outp;
    g.M = CC; g.Ntot = HWP; g.Ksplit = CC; g.bStrideK = HWP;
    g.bBatch = (long)CC * HWP; g.oBatch = (long)CC * HWP;
    gemm_wmma<2, 2, 1, 2 * CC><<<dim3((HWP + BN - 1) / BN, CC / BM, NB_), dim3(256), 0, stream>>>(g);
  }
}
